// Attention_34668976013567
// MI455X (gfx1250) — matrix operand order repaired and checked by reading, not compiled, not run on hardware
//
#include <hip/hip_runtime.h>
#include <hip/hip_bf16.h>

// ---------------------------------------------------------------------------
// Attention block for MI455X (gfx1250, wave32, WMMA f16->f32)
//   x[4,256,64,64] -> GroupNorm(32) -> QKV 1x1 -> 4-head attn (n=4096,d=32)
//   -> out 1x1 + bias
// Flash-attention tiling: the 16x[4096,4096] score matrix never touches HBM.
// GroupNorm is folded into the QKV weights (W' = W*s, qb = sum W*t).
// Attention waves use private LDS slices + wave_barrier (LDS is in-order
// within a wave on CDNA5) -- no block barriers in the hot loop.
// ---------------------------------------------------------------------------

typedef __attribute__((ext_vector_type(16))) _Float16 v16h;
typedef __attribute__((ext_vector_type(8)))  _Float16 v8h;
typedef __attribute__((ext_vector_type(8)))  float    v8f;

constexpr int kB   = 4;      // batch
constexpr int kC   = 256;    // channels
constexpr int kN   = 4096;   // spatial (64*64)
constexpr int kH   = 4;      // heads
constexpr int kF   = 32;     // dim per head
constexpr int kHID = 128;    // kH*kF
constexpr int kO3  = 384;    // 3*kHID
constexpr int kG   = 32;     // groupnorm groups
constexpr float kSCALE = 0.17677669529663687f; // 32^-0.5
constexpr float kEPS   = 1e-5f;

// ---- WMMA helpers ---------------------------------------------------------
__device__ inline v8f wmma16(v16h a, v16h b, v8f c) {
  // D = A(16x32 f16) * B(32x16 f16) + C(16x16 f32)
  return __builtin_amdgcn_wmma_f32_16x16x32_f16(
      /*neg_a=*/false, a, /*neg_b=*/false, b,
      /*c_mod=*/(short)0, c, /*reuse_a=*/false, /*reuse_b=*/false);
}

__device__ inline v8f vzero8() { v8f z = {0.f,0.f,0.f,0.f,0.f,0.f,0.f,0.f}; return z; }

// A-fragment (16x32 f16), contiguous k order matching B:
// lane L: m = L&15, g = L>>4; element e -> k = g*16 + e
// i.e. two contiguous 8-half runs at row offsets g*16 and g*16+8.
__device__ inline v16h load_a16(const _Float16* rowp, int g) {
  v8h lo = *(const v8h*)(rowp + g * 16);
  v8h hi = *(const v8h*)(rowp + g * 16 + 8);
  v16h r;
#pragma unroll
  for (int i = 0; i < 8; ++i) { r[i] = lo[i]; r[i + 8] = hi[i]; }
  return r;
}
// B-fragment (32x16 f16): lane L: n = L&15, g = L>>4; element e -> k = g*16+e.
// With our layouts each lane reads 16 contiguous halves (32B) -> plain v16h load.

// ---- Kernel 1: group statistics -> per-channel scale/shift ----------------
__global__ __launch_bounds__(256) void gn_stats_kernel(
    const float* __restrict__ x, const float* __restrict__ gw,
    const float* __restrict__ gb, float* __restrict__ sbuf,
    float* __restrict__ tbuf) {
  int blk = blockIdx.x;           // kB*kG blocks
  int b = blk >> 5, g = blk & 31; // 32 groups
  const int elems = (kC / kG) * kN;            // 8*4096 = 32768, contiguous
  const float4* p = (const float4*)(x + ((size_t)b * kC + g * 8) * kN);
  float s = 0.f, s2 = 0.f;
  for (int i = threadIdx.x; i < elems / 4; i += 256) {
    float4 v = p[i];
    s  += v.x + v.y + v.z + v.w;
    s2 += v.x * v.x + v.y * v.y + v.z * v.z + v.w * v.w;
  }
  __shared__ float r1[256], r2[256], bc[2];
  r1[threadIdx.x] = s; r2[threadIdx.x] = s2;
  __syncthreads();
  for (int st = 128; st > 0; st >>= 1) {
    if (threadIdx.x < st) {
      r1[threadIdx.x] += r1[threadIdx.x + st];
      r2[threadIdx.x] += r2[threadIdx.x + st];
    }
    __syncthreads();
  }
  if (threadIdx.x == 0) {
    float mean = r1[0] / (float)elems;
    float var  = r2[0] / (float)elems - mean * mean;
    bc[0] = mean;
    bc[1] = rsqrtf(var + kEPS);
  }
  __syncthreads();
  if (threadIdx.x < 8) {
    int c = g * 8 + threadIdx.x;
    float sc = bc[1] * gw[c];
    sbuf[b * kC + c] = sc;                      // xn = sc*x + sh
    tbuf[b * kC + c] = gb[c] - bc[0] * sc;
  }
}

// ---- Kernel 2: fold groupnorm into QKV weights ----------------------------
__global__ __launch_bounds__(256) void fold_w_kernel(
    const float* __restrict__ wqkv, const float* __restrict__ sbuf,
    const float* __restrict__ tbuf, _Float16* __restrict__ Ws,
    float* __restrict__ qb) {
  int b = blockIdx.x / kO3, o = blockIdx.x % kO3;
  int c = threadIdx.x;
  float w = wqkv[(size_t)o * kC + c];
  Ws[((size_t)b * kO3 + o) * kC + c] = (_Float16)(w * sbuf[b * kC + c]);
  __shared__ float red[256];
  red[c] = w * tbuf[b * kC + c];
  __syncthreads();
  for (int st = 128; st > 0; st >>= 1) {
    if (c < st) red[c] += red[c + st];
    __syncthreads();
  }
  if (c == 0) qb[b * kO3 + o] = red[0];
}

// ---- Kernel 3: QKV GEMM (384x256 @ 256xN), WMMA, LDS-staged x -------------
// Writes Q (pre-scaled by 1/sqrt(d)) and K as [b][h][n][f] f16,
// V transposed as [b][h][f][n] f16 (so attention B-frags are contiguous).
__global__ __launch_bounds__(256) void qkv_gemm_kernel(
    const float* __restrict__ x, const _Float16* __restrict__ Ws,
    const float* __restrict__ qb, _Float16* __restrict__ Qb,
    _Float16* __restrict__ Kb, _Float16* __restrict__ Vt) {
  constexpr int NB = 64;                       // n-columns per block
  int b = blockIdx.x / (kN / NB);
  int n0 = (blockIdx.x % (kN / NB)) * NB;
  int tid = threadIdx.x, lane = tid & 31, w = tid >> 5;
  int ln = lane & 15, g = lane >> 4;

  __shared__ float xs[32 * 65];                // 32 c-rows x 64 n-cols (+pad)
  __shared__ float qb_s[kO3];
  for (int i = tid; i < kO3; i += 256) qb_s[i] = qb[b * kO3 + i];

  v8f acc[12];
#pragma unroll
  for (int t = 0; t < 12; ++t) acc[t] = vzero8();

  for (int kc = 0; kc < kC; kc += 32) {
    __syncthreads();
    for (int i = tid; i < 32 * NB; i += 256) {
      int cc = i >> 6, nn = i & 63;
      xs[cc * 65 + nn] = x[((size_t)b * kC + kc + cc) * kN + n0 + nn];
    }
    __syncthreads();
    v16h af[3];
#pragma unroll
    for (int mi = 0; mi < 3; ++mi) {
      int o = (3 * w + mi) * 16 + ln;          // M row = output channel
      af[mi] = load_a16(Ws + ((size_t)b * kO3 + o) * kC + kc, g);
    }
    v16h bf[4];
#pragma unroll
    for (int ni = 0; ni < 4; ++ni) {
      int nl = ni * 16 + ln;
      v16h t;
#pragma unroll
      for (int e = 0; e < 16; ++e)
        t[e] = (_Float16)xs[(g * 16 + e) * 65 + nl];
      bf[ni] = t;
    }
#pragma unroll
    for (int mi = 0; mi < 3; ++mi)
#pragma unroll
      for (int ni = 0; ni < 4; ++ni)
        acc[mi * 4 + ni] = wmma16(af[mi], bf[ni], acc[mi * 4 + ni]);
  }

#pragma unroll
  for (int mi = 0; mi < 3; ++mi)
#pragma unroll
    for (int ni = 0; ni < 4; ++ni) {
      v8f a = acc[mi * 4 + ni];
      int n = n0 + ni * 16 + ln;
#pragma unroll
      for (int r = 0; r < 8; ++r) {
        int o = (3 * w + mi) * 16 + r + g * 8;
        float val = a[r] + qb_s[o];
        if (o < kHID) {
          int h = o >> 5, f = o & 31;
          Qb[(((size_t)b * kH + h) * kN + n) * kF + f] = (_Float16)(val * kSCALE);
        } else if (o < 2 * kHID) {
          int ol = o - kHID, h = ol >> 5, f = ol & 31;
          Kb[(((size_t)b * kH + h) * kN + n) * kF + f] = (_Float16)val;
        } else {
          int ol = o - 2 * kHID, h = ol >> 5, f = ol & 31;
          Vt[(((size_t)b * kH + h) * kF + f) * kN + n] = (_Float16)val;
        }
      }
    }
}

// ---- Kernel 4: flash attention --------------------------------------------
// One wave per 32-row query tile: 8 WMMAs per 32-key chunk, K/V fragments
// amortized over 2 row tiles. Online softmax: each lane owns exactly one
// query row (m/l running stats live in registers; only alpha crosses lanes
// via this wave's private LDS slice). No block barriers in the loop: LDS is
// in-order within a wave on CDNA5, wave_barrier() only pins compiler order.
__global__ __launch_bounds__(256) void attn_kernel(
    const _Float16* __restrict__ Qb, const _Float16* __restrict__ Kb,
    const _Float16* __restrict__ Vt, _Float16* __restrict__ AOt) {
  int tid = threadIdx.x, lane = tid & 31, w = tid >> 5;
  int wid = blockIdx.x * 8 + w;                // 2048 waves total
  int rt = wid & 127, h = (wid >> 7) & 3, b = wid >> 9;
  int row0 = rt * 32;
  int ln = lane & 15, g = lane >> 4;

  __shared__ float    S_s[8][32 * 33];         // stride 33: conflict-free
  __shared__ _Float16 P_s[8][32 * 32];
  __shared__ float    arow[8][32];
  __shared__ float    lrow[8][32];

  const _Float16* qbase = Qb + (((size_t)b * kH + h) * kN + row0) * kF;
  const _Float16* kbase = Kb + ((size_t)b * kH + h) * kN * kF;
  const _Float16* vbase = Vt + ((size_t)b * kH + h) * kF * kN;

  v16h qf0 = load_a16(qbase + (size_t)ln * kF, g);        // rows 0..15
  v16h qf1 = load_a16(qbase + (size_t)(16 + ln) * kF, g); // rows 16..31
  v8f acc00 = vzero8(), acc01 = vzero8();      // [rowtile][ftile]
  v8f acc10 = vzero8(), acc11 = vzero8();
  float m_run = -3.0e38f, l_run = 0.f;         // this lane's row = lane

  for (int kb = 0; kb < kN; kb += 32) {
    // K fragments: lane = key column, 32B contiguous each
    v16h k0 = *(const v16h*)(kbase + (size_t)(kb + ln) * kF + g * 16);
    v16h k1 = *(const v16h*)(kbase + (size_t)(kb + 16 + ln) * kF + g * 16);
    if (kb + 32 < kN)
      __builtin_prefetch(kbase + (size_t)(kb + 32 + ln) * kF, 0, 1);

    v8f s00 = wmma16(qf0, k0, vzero8());
    v8f s01 = wmma16(qf0, k1, vzero8());
    v8f s10 = wmma16(qf1, k0, vzero8());
    v8f s11 = wmma16(qf1, k1, vzero8());
#pragma unroll
    for (int r = 0; r < 8; ++r) {
      int m0 = (r + g * 8) * 33, m1 = (16 + r + g * 8) * 33;
      S_s[w][m0 + ln]      = s00[r];
      S_s[w][m0 + 16 + ln] = s01[r];
      S_s[w][m1 + ln]      = s10[r];
      S_s[w][m1 + 16 + ln] = s11[r];
    }
    __builtin_amdgcn_wave_barrier();           // compile-order fence only

    // online softmax: lane owns row == lane (EXEC stays all-ones)
    {
      const float* sr = &S_s[w][lane * 33];
      float mx = m_run;
#pragma unroll
      for (int c2 = 0; c2 < 32; ++c2) mx = fmaxf(mx, sr[c2]);
      float alpha = __expf(m_run - mx);
      float l = l_run * alpha;
#pragma unroll
      for (int c2 = 0; c2 < 32; ++c2) {
        float p = __expf(sr[c2] - mx);
        l += p;
        P_s[w][lane * 32 + c2] = (_Float16)p;
      }
      m_run = mx; l_run = l;
      arow[w][lane] = alpha;
    }
    __builtin_amdgcn_wave_barrier();

#pragma unroll
    for (int r = 0; r < 8; ++r) {              // rescale running output
      float a0 = arow[w][r + g * 8];
      float a1 = arow[w][16 + r + g * 8];
      acc00[r] *= a0; acc01[r] *= a0;
      acc10[r] *= a1; acc11[r] *= a1;
    }
    v16h pf0 = load_a16(&P_s[w][ln * 32], g);
    v16h pf1 = load_a16(&P_s[w][(16 + ln) * 32], g);
    v16h v0 = *(const v16h*)(vbase + (size_t)ln * kN + kb + g * 16);
    v16h v1 = *(const v16h*)(vbase + (size_t)(16 + ln) * kN + kb + g * 16);
    acc00 = wmma16(pf0, v0, acc00);            // rows 0..15, f 0..15
    acc01 = wmma16(pf0, v1, acc01);            // rows 0..15, f 16..31
    acc10 = wmma16(pf1, v0, acc10);            // rows 16..31, f 0..15
    acc11 = wmma16(pf1, v1, acc11);            // rows 16..31, f 16..31
    __builtin_amdgcn_wave_barrier();
  }

  lrow[w][lane] = l_run;
  __builtin_amdgcn_wave_barrier();
#pragma unroll
  for (int r = 0; r < 8; ++r) {
    float li0 = 1.f / lrow[w][r + g * 8];
    float li1 = 1.f / lrow[w][16 + r + g * 8];
    int n0r = row0 + r + g * 8;
    int n1r = row0 + 16 + r + g * 8;
    _Float16* d0 = AOt + ((size_t)b * kN + n0r) * kHID + h * kF;
    _Float16* d1 = AOt + ((size_t)b * kN + n1r) * kHID + h * kF;
    d0[ln]      = (_Float16)(acc00[r] * li0);
    d0[16 + ln] = (_Float16)(acc01[r] * li0);
    d1[ln]      = (_Float16)(acc10[r] * li1);
    d1[16 + ln] = (_Float16)(acc11[r] * li1);
  }
}

// ---- Kernel 5: output projection (256x128 @ 128xN) + bias -----------------
__global__ __launch_bounds__(256) void out_proj_kernel(
    const float* __restrict__ wout, const float* __restrict__ bout,
    const _Float16* __restrict__ AOt, float* __restrict__ y) {
  int tid = threadIdx.x, lane = tid & 31, w = tid >> 5;
  int wid = blockIdx.x * 8 + w;                 // 1024 waves
  int ng = wid & 15, mt = (wid >> 4) & 15, b = wid >> 8;
  int ln = lane & 15, g = lane >> 4;

  v16h af[4];
  int c = mt * 16 + ln;
#pragma unroll
  for (int kc = 0; kc < 4; ++kc) {
    v16h t;
#pragma unroll
    for (int e = 0; e < 16; ++e) {
      int k = g * 16 + e;
      t[e] = (_Float16)wout[(size_t)c * kHID + kc * 32 + k];
    }
    af[kc] = t;
  }
  float bo[8];
#pragma unroll
  for (int r = 0; r < 8; ++r) bo[r] = bout[mt * 16 + r + g * 8];

  for (int nt = ng * 16; nt < ng * 16 + 16; ++nt) {
    int n0 = nt * 16;
    v8f acc = vzero8();
#pragma unroll
    for (int kc = 0; kc < 4; ++kc) {
      v16h bf = *(const v16h*)(AOt + ((size_t)b * kN + n0 + ln) * kHID +
                               kc * 32 + g * 16);
      acc = wmma16(af[kc], bf, acc);
    }
#pragma unroll
    for (int r = 0; r < 8; ++r)
      y[((size_t)b * kC + mt * 16 + r + g * 8) * kN + n0 + ln] = acc[r] + bo[r];
  }
}

// ---- Launch ---------------------------------------------------------------
extern "C" void kernel_launch(void* const* d_in, const int* in_sizes, int n_in,
                              void* d_out, int out_size, void* d_ws,
                              size_t ws_size, hipStream_t stream) {
  (void)in_sizes; (void)n_in; (void)out_size; (void)ws_size;
  const float* x    = (const float*)d_in[0];
  const float* gw   = (const float*)d_in[1];
  const float* gb   = (const float*)d_in[2];
  const float* wqkv = (const float*)d_in[3];
  const float* wout = (const float*)d_in[4];
  const float* bout = (const float*)d_in[5];
  float* y = (float*)d_out;

  char* ws = (char*)d_ws;
  float* sbuf = (float*)(ws + 0);                              //  4 KB
  float* tbuf = (float*)(ws + 4096);                           //  4 KB
  float* qb   = (float*)(ws + 8192);                           //  6 KB
  _Float16* Ws = (_Float16*)(ws + 16384);                      // 768 KB
  _Float16* Qb = (_Float16*)(ws + 16384 + 786432);             //  4 MB each
  size_t qkvElems = (size_t)kB * kH * kN * kF;
  _Float16* Kb  = Qb + qkvElems;
  _Float16* Vt  = Kb + qkvElems;
  _Float16* AOt = Vt + qkvElems;

  gn_stats_kernel<<<kB * kG, 256, 0, stream>>>(x, gw, gb, sbuf, tbuf);
  fold_w_kernel<<<kB * kO3, 256, 0, stream>>>(wqkv, sbuf, tbuf, Ws, qb);
  qkv_gemm_kernel<<<kB * (kN / 64), 256, 0, stream>>>(x, Ws, qb, Qb, Kb, Vt);
  attn_kernel<<<(kB * kH * (kN / 32)) / 8, 256, 0, stream>>>(Qb, Kb, Vt, AOt);
  out_proj_kernel<<<(kB * 16 * 16) / 8, 256, 0, stream>>>(wout, bout, AOt, y);
}
